// NXTentLoss_65524021068077
// MI455X (gfx1250) — matrix operand order repaired and checked by reading, compiled, not run on hardware
//
#include <hip/hip_runtime.h>

typedef __attribute__((ext_vector_type(16))) _Float16 v16h;
typedef __attribute__((ext_vector_type(8)))  float    v8f;
typedef __attribute__((ext_vector_type(4)))  float    f4;
typedef __attribute__((ext_vector_type(4)))  _Float16 h4;
typedef int v4i __attribute__((__vector_size__(4 * sizeof(int))));

#define N2     8192
#define NHALF  4096
#define D      512
#define PITCH  520          // halfs per LDS row (512 + 8 pad -> 4-bank rotation)
#define MROWS  128          // rows per block (A tile)
#define JTILE  64           // cols per staged B tile (double buffered)
#define JSPLIT 8            // split of the 8192-column range across blocks
#define JCOLS  (N2 / JSPLIT)      // 1024 columns per block
#define NTILES (JCOLS / JTILE)    // 16 B tiles per block
#define INV_TEMP 2.0f

#if __has_builtin(__builtin_amdgcn_global_load_async_to_lds_b128)
#define HAVE_ASYNC_LDS 1
#else
#define HAVE_ASYNC_LDS 0
#endif

// Copy 16B global -> LDS. Async path uses the CDNA5 ASYNCcnt-tracked async
// load (GLOBAL_LOAD_ASYNC_TO_LDS_B128, no VGPR data movement); fallback is a
// plain load/store pair.
__device__ __forceinline__ void cp16_to_lds(const _Float16* g, _Float16* l) {
#if HAVE_ASYNC_LDS
  __builtin_amdgcn_global_load_async_to_lds_b128(
      (__attribute__((address_space(1))) v4i*)(const void*)g,
      (__attribute__((address_space(3))) v4i*)l, 0, 0);
#else
  *(f4*)l = *(const f4*)g;
#endif
}

__device__ __forceinline__ void wait_async_lds() {
  asm volatile("s_wait_asynccnt 0" ::: "memory");
}

// Stage `rows` rows (D halfs each) from xn[srcRow0..] into LDS at lbase.
__device__ __forceinline__ void stage_tile(const _Float16* __restrict__ xn,
                                           _Float16* lbase, int srcRow0,
                                           int rows, int tid) {
  const int chunks = rows * (D / 8);       // 16B chunks
  for (int c = tid; c < chunks; c += 256) {
    const int r  = c >> 6;
    const int k8 = (c & 63) << 3;
    cp16_to_lds(xn + (size_t)(srcRow0 + r) * D + k8,
                lbase + (size_t)r * PITCH + k8);
  }
}

// ---------------------------------------------------------------------------
// Kernel 1: row L2-normalize (f32) and cast to f16 into workspace.
// ---------------------------------------------------------------------------
__global__ __launch_bounds__(128) void nxt_normalize(const float* __restrict__ x,
                                                     _Float16* __restrict__ xn) {
  __shared__ float wpart[4];
  const int row = blockIdx.x;
  const int tid = threadIdx.x;

  f4 v = *(const f4*)(x + (size_t)row * D + tid * 4);
  float s = v[0]*v[0] + v[1]*v[1] + v[2]*v[2] + v[3]*v[3];
  #pragma unroll
  for (int m = 1; m < 32; m <<= 1) s += __shfl_xor(s, m, 32);
  if ((tid & 31) == 0) wpart[tid >> 5] = s;
  __syncthreads();
  const float tot   = wpart[0] + wpart[1] + wpart[2] + wpart[3];
  const float scale = 1.0f / fmaxf(sqrtf(tot), 1e-6f);

  h4 h;
  h[0] = (_Float16)(v[0] * scale);
  h[1] = (_Float16)(v[1] * scale);
  h[2] = (_Float16)(v[2] * scale);
  h[3] = (_Float16)(v[3] * scale);
  *(h4*)(xn + (size_t)row * D + tid * 4) = h;
}

// ---------------------------------------------------------------------------
// Kernel 2: fused GEMM (Xn * Xn^T, f16 WMMA, f32 acc) + per-row sum(exp) and
// positive-pair extraction. Each block: 128 rows x 1024 cols, K = 512.
// B tiles are double-buffered with async global->LDS prefetch.
// ---------------------------------------------------------------------------
__global__ __launch_bounds__(256) void nxt_main(const _Float16* __restrict__ xn,
                                                float* __restrict__ rowsum,
                                                float* __restrict__ possum) {
  extern __shared__ __align__(16) char smem_raw[];
  _Float16* Ab  = (_Float16*)smem_raw;                   // MROWS * PITCH halfs
  _Float16* Bb0 = Ab  + (size_t)MROWS * PITCH;           // JTILE * PITCH halfs
  _Float16* Bb1 = Bb0 + (size_t)JTILE * PITCH;           // JTILE * PITCH halfs

  const int mb      = blockIdx.x & ((N2 / MROWS) - 1);   // 0..63
  const int js      = blockIdx.x / (N2 / MROWS);         // 0..JSPLIT-1
  const int rowBase = mb * MROWS;
  const int jOrigin = js * JCOLS;

  const int tid  = threadIdx.x;
  const int lane = tid & 31;
  const int w    = tid >> 5;       // wave 0..7 -> 16-row slice
  const int nl   = lane & 15;      // column-within-tile / row-within-frag
  const int grp  = lane >> 4;      // half-wave group

  // Prologue: stage A (once) and B tile 0 into buf0.
  stage_tile(xn, Ab,  rowBase, MROWS, tid);
  stage_tile(xn, Bb0, jOrigin, JTILE, tid);
  wait_async_lds();
  __syncthreads();

  float rs[8] = {};   // running sum(exp(logit)) per row (lane-partial over cols)
  float ps[8] = {};   // positive-pair logit per row (exactly one contributor)
  const int waveRow = rowBase + w * 16;

  for (int jb = 0; jb < NTILES; ++jb) {
    _Float16* cur = (jb & 1) ? Bb1 : Bb0;
    _Float16* nxt = (jb & 1) ? Bb0 : Bb1;

    // Prefetch next B tile while computing on the current one.
    if (jb + 1 < NTILES)
      stage_tile(xn, nxt, jOrigin + (jb + 1) * JTILE, JTILE, tid);

    v8f acc[4] = {};
    for (int kc = 0; kc < D / 32; ++kc) {
      // A fragment: 16-bit A 16x32 layout -> chunks at K = grp*8 and grp*8+16
      const _Float16* ap = Ab + (size_t)(w * 16 + nl) * PITCH + kc * 32 + grp * 8;
      v16h a;
      *(f4*)&a       = *(const f4*)ap;
      *((f4*)&a + 1) = *(const f4*)(ap + 16);

      // Load ALL B fragments first (distinct regs) so the scheduler can issue
      // descending s_wait_dscnt instead of serializing each WMMA on dscnt==0.
      v16h bf[4];
      #pragma unroll
      for (int jt = 0; jt < 4; ++jt) {
        const _Float16* bp =
            cur + (size_t)(jt * 16 + nl) * PITCH + kc * 32 + grp * 8;
        *(f4*)&bf[jt]       = *(const f4*)bp;
        *((f4*)&bf[jt] + 1) = *(const f4*)(bp + 16);
      }
      #pragma unroll
      for (int jt = 0; jt < 4; ++jt)
        acc[jt] = __builtin_amdgcn_wmma_f32_16x16x32_f16(
            false, a, false, bf[jt], (short)0, acc[jt], false, false);
    }

    // Post-process: logits bounded by |2| -> no max-tracking needed.
    const int jBase = jOrigin + jb * JTILE;
    #pragma unroll
    for (int jt = 0; jt < 4; ++jt) {
      const int col = jBase + jt * 16 + nl;
      #pragma unroll
      for (int v = 0; v < 8; ++v) {
        const int row = waveRow + grp * 8 + v;     // C/D layout: M = v + grp*8
        const float l = acc[jt][v] * INV_TEMP;
        float e = __expf(l);
        e = (col == row) ? 0.0f : e;               // == -inf diagonal mask
        rs[v] += e;
        const int pos = (row < NHALF) ? row + NHALF : row - NHALF;
        ps[v] += (col == pos) ? l : 0.0f;
      }
    }

    wait_async_lds();   // next tile fully in LDS (own wave's share)
    __syncthreads();    // all waves done: reads of cur finished, nxt visible
  }

  // Butterfly reduce across the 16 lanes of each half-wave group.
  #pragma unroll
  for (int v = 0; v < 8; ++v) {
    #pragma unroll
    for (int m = 1; m <= 8; m <<= 1) {
      rs[v] += __shfl_xor(rs[v], m, 32);
      ps[v] += __shfl_xor(ps[v], m, 32);
    }
  }
  if (nl == 0) {
    #pragma unroll
    for (int v = 0; v < 8; ++v) {
      const int row = waveRow + grp * 8 + v;
      rowsum[(size_t)js * N2 + row] = rs[v];
      possum[(size_t)js * N2 + row] = ps[v];
    }
  }
}

// ---------------------------------------------------------------------------
// Kernel 3: deterministic finalize. loss = sum_i( log(sum_exp_i) - pos_i ).
// ---------------------------------------------------------------------------
__global__ __launch_bounds__(256) void nxt_finalize(const float* __restrict__ rowsum,
                                                    const float* __restrict__ possum,
                                                    float* __restrict__ out) {
  __shared__ float part[256];
  const int tid = threadIdx.x;
  float acc = 0.0f;
  for (int r = tid; r < N2; r += 256) {
    float s = 0.0f, p = 0.0f;
    #pragma unroll
    for (int q = 0; q < JSPLIT; ++q) {
      s += rowsum[(size_t)q * N2 + r];
      p += possum[(size_t)q * N2 + r];
    }
    acc += logf(s) - p;
  }
  part[tid] = acc;
  __syncthreads();
  for (int off = 128; off > 0; off >>= 1) {
    if (tid < off) part[tid] += part[tid + off];
    __syncthreads();
  }
  if (tid == 0) *out = part[0];
}

// ---------------------------------------------------------------------------
extern "C" void kernel_launch(void* const* d_in, const int* in_sizes, int n_in,
                              void* d_out, int out_size, void* d_ws, size_t ws_size,
                              hipStream_t stream) {
  (void)in_sizes; (void)n_in; (void)out_size; (void)ws_size;
  const float* x   = (const float*)d_in[0];
  float*       out = (float*)d_out;

  _Float16* xn     = (_Float16*)d_ws;                                   // 8 MB
  float*    rowsum = (float*)((char*)d_ws + (size_t)N2 * D * sizeof(_Float16));
  float*    possum = rowsum + (size_t)JSPLIT * N2;                      // +256KB

  nxt_normalize<<<N2, 128, 0, stream>>>(x, xn);

  const size_t ldsBytes =
      (size_t)(MROWS + 2 * JTILE) * PITCH * sizeof(_Float16);           // 260 KB
  nxt_main<<<(N2 / MROWS) * JSPLIT, 256, ldsBytes, stream>>>(xn, rowsum, possum);

  nxt_finalize<<<1, 256, 0, stream>>>(rowsum, possum, out);
}
